// GraphAttentionLayer_2010044695310
// MI455X (gfx1250) — matrix operand order repaired and checked by reading, not compiled, not run on hardware
//
#include <hip/hip_runtime.h>

#define N_NODES 12288
#define IN_DIM  128
#define OUT_DIM 64
#define LRALPHA 0.2f

typedef __attribute__((ext_vector_type(16))) _Float16 v16h;
typedef __attribute__((ext_vector_type(8)))  float    v8f;

// ---------------------------------------------------------------------------
// Phase 1: per-row GEMV block. Computes Wh[row,:] = h[row,:] @ W, stores it
// transposed in fp16 (WhT[col][row]) for WMMA B-fragments, and the two
// attention projections Wh1[row] = Wh.a1, Wh2[row] = Wh.a2.
// ---------------------------------------------------------------------------
__global__ __launch_bounds__(64)
void gat_phase1(const float* __restrict__ h, const float* __restrict__ W,
                const float* __restrict__ a, _Float16* __restrict__ WhT,
                float* __restrict__ Wh1, float* __restrict__ Wh2) {
  const int row = blockIdx.x;
  const int c   = threadIdx.x;          // output column 0..63
  __shared__ float sh[IN_DIM];
  sh[c]      = h[(size_t)row * IN_DIM + c];
  sh[c + 64] = h[(size_t)row * IN_DIM + 64 + c];
  __syncthreads();
  float acc = 0.f;
#pragma unroll 8
  for (int k = 0; k < IN_DIM; ++k)
    acc = fmaf(sh[k], W[k * OUT_DIM + c], acc);
  WhT[(size_t)c * N_NODES + row] = (_Float16)acc;

  __shared__ float s1[64], s2[64];
  s1[c] = acc * a[c];
  s2[c] = acc * a[OUT_DIM + c];
  __syncthreads();
  for (int s = 32; s > 0; s >>= 1) {
    if (c < s) { s1[c] += s1[c + s]; s2[c] += s2[c + s]; }
    __syncthreads();
  }
  if (c == 0) { Wh1[row] = s1[0]; Wh2[row] = s2[0]; }
}

// ---------------------------------------------------------------------------
// Phase 2: global max of Wh2 (rank-1 score matrix => closed-form softmax max).
// ---------------------------------------------------------------------------
__global__ __launch_bounds__(256)
void gat_max(const float* __restrict__ Wh2, float* __restrict__ M2) {
  __shared__ float sm[256];
  float m = -3.402823466e38f;
  for (int i = threadIdx.x; i < N_NODES; i += 256) m = fmaxf(m, Wh2[i]);
  sm[threadIdx.x] = m;
  __syncthreads();
  for (int s = 128; s > 0; s >>= 1) {
    if (threadIdx.x < s)
      sm[threadIdx.x] = fmaxf(sm[threadIdx.x], sm[threadIdx.x + s]);
    __syncthreads();
  }
  if (threadIdx.x == 0) *M2 = sm[0];
}

// ---------------------------------------------------------------------------
// Phase 3: fused softmax(leakyrelu(Wh1_i + Wh2_j)) @ Wh, then ELU.
// One wave (32 lanes) owns a 16-row output tile; 4 waves per block.
// Per 32-wide K step: build the 16x32 fp16 A fragment (probabilities) in
// registers and issue 4x v_wmma_f32_16x16x32_f16 against Whᵀ columns.
// ---------------------------------------------------------------------------
__global__ __launch_bounds__(128)
void gat_attn(const _Float16* __restrict__ WhT, const float* __restrict__ Wh1,
              const float* __restrict__ Wh2, const float* __restrict__ M2,
              float* __restrict__ out) {
  __shared__ float sWh2[N_NODES];      // 48 KB: all attention keys
  __shared__ float sZ[4][32];

  const int tid = threadIdx.x;
  for (int i = tid; i < N_NODES; i += 128) sWh2[i] = Wh2[i];

  const int wv   = tid >> 5;
  const int lane = tid & 31;
  const int m    = lane & 15;          // A-fragment row / D-fragment column
  const int kh   = lane >> 4;          // K half-select (ISA 16-bit layouts)
  const int i0   = (blockIdx.x * 4 + wv) * 16;

  const float r1 = Wh1[i0 + m];
  const float m2 = *M2;
  float mi = r1 + m2;                  // row max = lrelu(Wh1_i + max_j Wh2_j)
  mi = (mi >= 0.f) ? mi : LRALPHA * mi;
  __syncthreads();

  // B fragments: lane = column n (= m), 16 contiguous K halves at 16*kh.
  const _Float16* bp0 = WhT + (size_t)(m + 0)  * N_NODES + 16 * kh;
  const _Float16* bp1 = bp0 + (size_t)16 * N_NODES;
  const _Float16* bp2 = bp0 + (size_t)32 * N_NODES;
  const _Float16* bp3 = bp0 + (size_t)48 * N_NODES;

  v8f acc0 = {}, acc1 = {}, acc2 = {}, acc3 = {};
  float zpart = 0.f;

  for (int j0 = 0; j0 < N_NODES; j0 += 32) {
    // A-fragment K map (contiguous, matches B): t -> K=16*kh+t
    const float* w2 = &sWh2[j0 + 16 * kh];
    float4 t0 = *(const float4*)(w2);
    float4 t1 = *(const float4*)(w2 + 4);
    float4 t2 = *(const float4*)(w2 + 8);
    float4 t3 = *(const float4*)(w2 + 12);
    float vals[16] = { t0.x, t0.y, t0.z, t0.w,  t1.x, t1.y, t1.z, t1.w,
                       t2.x, t2.y, t2.z, t2.w,  t3.x, t3.y, t3.z, t3.w };
    v16h aF;
#pragma unroll
    for (int t = 0; t < 16; ++t) {
      float e = r1 + vals[t];
      e = (e >= 0.f) ? e : LRALPHA * e;
      float p = __expf(e - mi);        // in (0,1]: fp16-safe
      zpart += p;
      aF[t] = (_Float16)p;
    }
    v16h bF0 = *(const v16h*)(bp0 + j0);
    v16h bF1 = *(const v16h*)(bp1 + j0);
    v16h bF2 = *(const v16h*)(bp2 + j0);
    v16h bF3 = *(const v16h*)(bp3 + j0);
    acc0 = __builtin_amdgcn_wmma_f32_16x16x32_f16(false, aF, false, bF0, (short)0, acc0, false, false);
    acc1 = __builtin_amdgcn_wmma_f32_16x16x32_f16(false, aF, false, bF1, (short)0, acc1, false, false);
    acc2 = __builtin_amdgcn_wmma_f32_16x16x32_f16(false, aF, false, bF2, (short)0, acc2, false, false);
    acc3 = __builtin_amdgcn_wmma_f32_16x16x32_f16(false, aF, false, bF3, (short)0, acc3, false, false);
  }

  sZ[wv][lane] = zpart;                // lanes (m, kh=0/1) hold disjoint halves
  __syncthreads();

  float zinv[8];
#pragma unroll
  for (int v = 0; v < 8; ++v) {
    int r = v + 8 * kh;                // D-fragment row for VGPR v
    zinv[v] = 1.f / (sZ[wv][r] + sZ[wv][r + 16]);
  }
#pragma unroll
  for (int v = 0; v < 8; ++v) {
    int row = i0 + v + 8 * kh;
    float x;
    x = acc0[v] * zinv[v]; out[row * OUT_DIM +  0 + m] = (x > 0.f) ? x : (__expf(x) - 1.f);
    x = acc1[v] * zinv[v]; out[row * OUT_DIM + 16 + m] = (x > 0.f) ? x : (__expf(x) - 1.f);
    x = acc2[v] * zinv[v]; out[row * OUT_DIM + 32 + m] = (x > 0.f) ? x : (__expf(x) - 1.f);
    x = acc3[v] * zinv[v]; out[row * OUT_DIM + 48 + m] = (x > 0.f) ? x : (__expf(x) - 1.f);
  }
}

// ---------------------------------------------------------------------------
extern "C" void kernel_launch(void* const* d_in, const int* in_sizes, int n_in,
                              void* d_out, int out_size, void* d_ws, size_t ws_size,
                              hipStream_t stream) {
  const float* h = (const float*)d_in[0];   // [N, 128]
  const float* W = (const float*)d_in[1];   // [128, 64]
  const float* a = (const float*)d_in[2];   // [128, 1]
  float* out = (float*)d_out;               // [N, 64]

  char* ws = (char*)d_ws;
  _Float16* WhT = (_Float16*)ws;                                   // 64*N fp16 = 1.5 MB
  float* Wh1 = (float*)(ws + (size_t)OUT_DIM * N_NODES * sizeof(_Float16));
  float* Wh2 = Wh1 + N_NODES;
  float* M2  = Wh2 + N_NODES;

  gat_phase1<<<N_NODES, 64, 0, stream>>>(h, W, a, WhT, Wh1, Wh2);
  gat_max<<<1, 256, 0, stream>>>(Wh2, M2);
  gat_attn<<<N_NODES / 64, 128, 0, stream>>>(WhT, Wh1, Wh2, M2, out);
}
